// PyTorchMLP_14723147891335
// MI455X (gfx1250) — matrix operand order repaired and checked by reading, not compiled, not run on hardware
//
#include <hip/hip_runtime.h>

// Fused MLP: o = silu(x @ w1) @ w2 per head. H=16, N=8192, D=256, I=1024.
// f32 in HBM; bf16 WMMA math (v_wmma_f32_16x16x32_bf16), f32 accumulate.
// 512 threads = 16 waves. Per wave: MT=2 row tiles x NT=4 col tiles (B fragment
// reused across 2 WMMAs, A across 4). Weight slabs [32k x 64col] staged by wave
// pairs into LDS (transposed, bf16, packed cvt + b64 stores), double buffered.
// Phase 2 splits K across wave halves; partials reduced via LDS scratch.

typedef __attribute__((ext_vector_type(16))) __bf16 v16bf;
typedef __attribute__((ext_vector_type(8)))  __bf16 v8bf;
typedef __attribute__((ext_vector_type(4)))  __bf16 v4bf;
typedef __attribute__((ext_vector_type(8)))  float  v8f;
typedef __attribute__((ext_vector_type(4)))  float  v4f;

constexpr int kH = 16;
constexpr int kN = 8192;
constexpr int kD = 256;
constexpr int kI = 1024;

constexpr int BM = 64;                    // rows per workgroup
constexpr int XS_STRIDE = kD + 8;         // 264 bf16
constexpr int HS_STRIDE = kI + 8;         // 1032 bf16 (bank-conflict-free b128)
constexpr int WB_STRIDE = 40;             // bf16 per slab column (32 k + pad)
constexpr int WSLAB     = 64 * WB_STRIDE; // [64col][32k] slab, elems
// LDS: Xs + Hs + 8 groups x 2 buffers of slab (slabs double as f32 scratch)
constexpr int SMEM_BYTES =
    (BM * XS_STRIDE + BM * HS_STRIDE + 8 * 2 * WSLAB) * 2;   // 247,808 B

// ---- fragment loaders ------------------------------------------------------

// A fragment (16x32 bf16) from LDS tile; ISA 7.12.2 16-bit A layout.
__device__ __forceinline__ v16bf load_A_lds(const __bf16* base, int stride,
                                            int mrow0, int kk, int lane) {
    const __bf16* p = base + (mrow0 + (lane & 15)) * stride + kk + ((lane >> 4) << 3);
    v8bf lo = *(const v8bf*)(p);                      // ds_load_b128
    v8bf hi = *(const v8bf*)(p + 16);                 // ds_load_b128
    return __builtin_shufflevector(lo, hi, 0,1,2,3,4,5,6,7,8,9,10,11,12,13,14,15);
}

// B fragment (32x16 bf16) from transposed slab [col][k]; same K order as A.
__device__ __forceinline__ v16bf load_B_lds(const __bf16* slab, int t, int lane) {
    const __bf16* p = slab + (t * 16 + (lane & 15)) * WB_STRIDE + ((lane >> 4) << 3);
    v8bf lo = *(const v8bf*)(p);                      // ds_load_b128
    v8bf hi = *(const v8bf*)(p + 16);                 // ds_load_b128
    return __builtin_shufflevector(lo, hi, 0,1,2,3,4,5,6,7,8,9,10,11,12,13,14,15);
}

// Stage W[kk..kk+32)[c0..c0+64) (f32, ld=ldb) -> transposed bf16 slab.
// 64 threads (one wave pair). Task = 4x4 (k-quad x col-quad): packed converts
// + one ds_store_b64 per column.
__device__ __forceinline__ void stage_slab(const float* __restrict__ Wg, int ldb,
                                           int kk, int c0, __bf16* slab, int ptid) {
#pragma unroll
    for (int i = 0; i < 2; ++i) {
        int task = ptid + i * 64;                     // 0..127
        int kq   = (task >> 4) << 2;                  // k 0,4,..,28
        int j4   = (task & 15) << 2;                  // col 0,4,..,60
        const float* p = Wg + (size_t)(kk + kq) * ldb + c0 + j4;
        v4f d0 = *(const v4f*)(p);                    // global_load_b128 x4
        v4f d1 = *(const v4f*)(p + ldb);
        v4f d2 = *(const v4f*)(p + 2 * (size_t)ldb);
        v4f d3 = *(const v4f*)(p + 3 * (size_t)ldb);
#pragma unroll
        for (int e = 0; e < 4; ++e) {
            v4bf b;
            b[0] = (__bf16)d0[e]; b[1] = (__bf16)d1[e];   // v_cvt_pk_bf16_f32
            b[2] = (__bf16)d2[e]; b[3] = (__bf16)d3[e];   // v_cvt_pk_bf16_f32
            *(v4bf*)(&slab[(j4 + e) * WB_STRIDE + kq]) = b;  // ds_store_b64
        }
    }
}

__device__ __forceinline__ v8f wmma_bf16(v16bf A, v16bf B, v8f C) {
    return __builtin_amdgcn_wmma_f32_16x16x32_bf16(false, A, false, B,
                                                   (short)0, C, false, false);
}

// One k-step: 2 A fragments, 4 B fragments, 8 WMMAs.
__device__ __forceinline__ void compute_step(const __bf16* Asrc, int astride,
                                             int mi0, int kk, const __bf16* cur,
                                             int lane, v8f acc[8]) {
    v16bf A0 = load_A_lds(Asrc, astride, mi0 * 16, kk, lane);
    v16bf A1 = load_A_lds(Asrc, astride, mi0 * 16 + 16, kk, lane);
#pragma unroll
    for (int t = 0; t < 4; ++t) {
        v16bf B = load_B_lds(cur, t, lane);
        acc[t]     = wmma_bf16(A0, B, acc[t]);
        acc[4 + t] = wmma_bf16(A1, B, acc[4 + t]);
    }
}

// ---- kernel ----------------------------------------------------------------

__global__ __launch_bounds__(512) void
fused_mlp_bf16_wmma(const float* __restrict__ x,
                    const float* __restrict__ w1,
                    const float* __restrict__ w2,
                    float* __restrict__ out) {
    extern __shared__ char smem_raw[];
    __bf16* Xs   = (__bf16*)smem_raw;                 // [BM][XS_STRIDE]
    __bf16* Hs   = Xs + BM * XS_STRIDE;               // [BM][HS_STRIDE]
    __bf16* Wbuf = Hs + BM * HS_STRIDE;               // [8 grp][2 buf][WSLAB]

    const int rb   = blockIdx.x;
    const int head = blockIdx.y;
    const int tid  = threadIdx.x;
    const int wave = tid >> 5;                        // 0..15
    const int lane = tid & 31;
    const int mi0  = (wave & 1) * 2;                  // first of 2 row tiles
    const int grp  = wave >> 1;                       // 0..7 (wave pair)
    const int ptid = tid & 63;                        // id within wave pair
    __bf16* Wp = Wbuf + grp * 2 * WSLAB;              // this pair's 2 slabs

    const float* xh  = x   + ((size_t)head * kN + (size_t)rb * BM) * kD;
    const float* w1h = w1  + (size_t)head * kD * kI;
    const float* w2h = w2  + (size_t)head * kI * kD;
    float*       oh  = out + ((size_t)head * kN + (size_t)rb * BM) * kD;

    // ---- stage X tile (64x256 f32 -> bf16 LDS); prefetch w2 head -----------
#pragma unroll
    for (int i = 0; i < 2; ++i)
        __builtin_prefetch((const char*)w2h + ((size_t)tid * 2 + i) * 1024, 0, 0);
#pragma unroll
    for (int i = 0; i < 8; ++i) {
        int idx = tid + i * 512;
        int e4  = idx << 2;
        int row = e4 >> 8;
        int col = e4 & 255;
        v4f d = *(const v4f*)(xh + (size_t)row * kD + col);
        v4bf b;
#pragma unroll
        for (int j = 0; j < 4; ++j) b[j] = (__bf16)d[j];
        *(v4bf*)(&Xs[row * XS_STRIDE + col]) = b;
    }
    __syncthreads();

    // ---- Phase 1: Z = X @ W1, SiLU -> Hs (bf16) ----------------------------
    // Group grp covers cols g*512 + grp*64 for g in {0,1}; K = 256, 8 k-steps.
#pragma unroll 1
    for (int g = 0; g < 2; ++g) {
        const int c0 = g * 512 + grp * 64;
        v8f acc[8] = {};
        stage_slab(w1h, kI, 0, c0, Wp, ptid);
        __syncthreads();
#pragma unroll 2
        for (int ks = 0; ks < 8; ++ks) {
            const int kk = ks * 32;
            __bf16* cur = Wp + ((ks & 1) ? WSLAB : 0);
            __bf16* nxt = Wp + ((ks & 1) ? 0 : WSLAB);
            if (ks + 1 < 8) stage_slab(w1h, kI, kk + 32, c0, nxt, ptid);
            compute_step(Xs, XS_STRIDE, mi0, kk, cur, lane, acc);
            __syncthreads();
        }
        // SiLU + bf16 -> Hs. C/D layout: VGPR v -> row r0+v, lane&15 -> col.
#pragma unroll
        for (int mt = 0; mt < 2; ++mt) {
            const int r0 = (mi0 + mt) * 16 + ((lane >> 4) << 3);
#pragma unroll
            for (int t = 0; t < 4; ++t) {
                const int col = c0 + t * 16 + (lane & 15);
#pragma unroll
                for (int v = 0; v < 8; ++v) {
                    float z = acc[mt * 4 + t][v];
                    float s = z * __frcp_rn(1.0f + __expf(-z));   // silu
                    Hs[(r0 + v) * HS_STRIDE + col] = (__bf16)s;
                }
            }
        }
    }
    __syncthreads();

    // ---- Phase 2: O = Hs @ W2, K split across wave halves ------------------
    // grp -> (cg = grp&3 -> cols cg*64, kh = grp>>2 -> K half), 16 k-steps.
    {
        const int cg    = grp & 3;
        const int kh    = grp >> 2;
        const int c0    = cg * 64;
        const int kbase = kh * 512;
        v8f acc[8] = {};
        stage_slab(w2h, kD, kbase, c0, Wp, ptid);
        __syncthreads();
#pragma unroll 2
        for (int ks = 0; ks < 16; ++ks) {
            const int kk = kbase + ks * 32;
            __bf16* cur = Wp + ((ks & 1) ? WSLAB : 0);
            __bf16* nxt = Wp + ((ks & 1) ? 0 : WSLAB);
            if (ks + 1 < 16) stage_slab(w2h, kD, kk + 32, c0, nxt, ptid);
            compute_step(Hs, HS_STRIDE, mi0, kk, cur, lane, acc);
            __syncthreads();
        }
        // Reduce the two K halves through f32 LDS scratch (reuses slab space).
        float* scratch = (float*)Wbuf;                // 8 slots x 2048 f32
        const int slot = (wave & 7) * 2048;
        if (kh == 1) {
#pragma unroll
            for (int i = 0; i < 8; ++i)
                *(v8f*)(&scratch[slot + i * 256 + lane * 8]) = acc[i];
        }
        __syncthreads();
        if (kh == 0) {
#pragma unroll
            for (int i = 0; i < 8; ++i) {
                v8f part = *(const v8f*)(&scratch[slot + i * 256 + lane * 8]);
#pragma unroll
                for (int v = 0; v < 8; ++v) acc[i][v] += part[v];
            }
#pragma unroll
            for (int mt = 0; mt < 2; ++mt) {
                const int r0 = (mi0 + mt) * 16 + ((lane >> 4) << 3);
#pragma unroll
                for (int t = 0; t < 4; ++t) {
                    const int col = c0 + t * 16 + (lane & 15);
#pragma unroll
                    for (int v = 0; v < 8; ++v)
                        oh[(size_t)(r0 + v) * kD + col] = acc[mt * 4 + t][v];
                }
            }
        }
    }
}

// NOTE on scratch exchange layout: the v8f store puts lane-local acc rows at
// scratch[slot + i*256 + lane*8 .. +7]; the kh=0 partner reads the identical
// addresses, so the pairwise sum is exact regardless of the WMMA lane mapping.

// ---- launch ----------------------------------------------------------------

extern "C" void kernel_launch(void* const* d_in, const int* in_sizes, int n_in,
                              void* d_out, int out_size, void* d_ws, size_t ws_size,
                              hipStream_t stream) {
    const float* x  = (const float*)d_in[0];
    const float* w1 = (const float*)d_in[1];
    const float* w2 = (const float*)d_in[2];
    float* out = (float*)d_out;

    dim3 grid(kN / BM, kH);      // (128, 16)
    dim3 block(512);             // 16 waves (wave32)
    fused_mlp_bf16_wmma<<<grid, block, SMEM_BYTES, stream>>>(x, w1, w2, out);
}
